// SimpleVQVAE_11476152615192
// MI455X (gfx1250) — matrix operand order repaired and checked by reading, not compiled, not run on hardware
//
#include <hip/hip_runtime.h>
#include <hip/hip_bf16.h>

typedef __bf16 bf16;
typedef __attribute__((ext_vector_type(16))) __bf16 v16bf;
typedef __attribute__((ext_vector_type(8)))  __bf16 v8bf;
typedef __attribute__((ext_vector_type(8)))  float  v8f;

#define D_MODEL 768
#define HID     128
#define CD      64
#define NCODE   2048
#define BN      65536   // 16 * 4096 tokens

// ---------------------------------------------------------------- helpers

__device__ __forceinline__ float gelu_exact(float v) {
    return 0.5f * v * (1.0f + erff(v * 0.70710678118654752f));
}

__device__ __forceinline__ v8f v8f_zero() {
    v8f z;
#pragma unroll
    for (int i = 0; i < 8; i++) z[i] = 0.f;
    return z;
}

// A-fragment (16x32 bf16, ISA 7.12.2 layout) built from fp32 rows.
// p points at row_base + k0 + (lane>=16 ? 8 : 0); loads K..K+7 and K+16..K+23.
__device__ __forceinline__ v16bf load_a_from_f32(const float* p) {
    float4 f0 = *(const float4*)(p);
    float4 f1 = *(const float4*)(p + 4);
    float4 f2 = *(const float4*)(p + 16);
    float4 f3 = *(const float4*)(p + 20);
    v16bf a;
    a[0]=(bf16)f0.x;  a[1]=(bf16)f0.y;  a[2]=(bf16)f0.z;  a[3]=(bf16)f0.w;
    a[4]=(bf16)f1.x;  a[5]=(bf16)f1.y;  a[6]=(bf16)f1.z;  a[7]=(bf16)f1.w;
    a[8]=(bf16)f2.x;  a[9]=(bf16)f2.y;  a[10]=(bf16)f2.z; a[11]=(bf16)f2.w;
    a[12]=(bf16)f3.x; a[13]=(bf16)f3.y; a[14]=(bf16)f3.z; a[15]=(bf16)f3.w;
    return a;
}

// Same A-fragment from bf16 storage (global or LDS), two 16-byte loads.
__device__ __forceinline__ v16bf load_a_from_bf16(const bf16* p) {
    v8bf lo  = *(const v8bf*)(p);
    v8bf hi8 = *(const v8bf*)(p + 16);
    v16bf a;
#pragma unroll
    for (int i = 0; i < 8; i++) { a[i] = lo[i]; a[8 + i] = hi8[i]; }
    return a;
}

__device__ __forceinline__ v8f wmma_bf16(v16bf a, v16bf b, v8f c) {
    // 8 args: (neg_a, A, neg_b, B, c_mod, C, reuse_a, reuse_b)
    return __builtin_amdgcn_wmma_f32_16x16x32_bf16(false, a, false, b,
                                                   (short)0, c, false, false);
}

// ---------------------------------------------------------------- prep:
// transpose/convert weights to bf16 (N,K) layout, codebook norms, zero loss.
__global__ void vq_prep(const float* __restrict__ w1, const float* __restrict__ w2,
                        const float* __restrict__ cb,
                        const float* __restrict__ d1, const float* __restrict__ d2,
                        bf16* __restrict__ w1t, bf16* __restrict__ w2t,
                        bf16* __restrict__ cbb, float* __restrict__ cnorm,
                        bf16* __restrict__ d1t, bf16* __restrict__ d2t,
                        float* __restrict__ out_loss) {
    int tid = blockIdx.x * blockDim.x + threadIdx.x;
    if (tid < HID * D_MODEL) {               // enc_w1 (768,128) -> (128,768)
        int n = tid / D_MODEL, k = tid % D_MODEL;
        w1t[tid] = (bf16)w1[k * HID + n];
    }
    if (tid < CD * HID) {                    // enc_w2 (128,64) -> (64,128)
        int n = tid / HID, k = tid % HID;
        w2t[tid] = (bf16)w2[k * CD + n];
    }
    if (tid < NCODE * CD) cbb[tid] = (bf16)cb[tid];   // codebook already (N,K)
    if (tid < NCODE) {
        float s = 0.f;
        for (int k = 0; k < CD; k++) { float v = cb[tid * CD + k]; s += v * v; }
        cnorm[tid] = s;
    }
    if (tid < HID * CD) {                    // dec_w1 (64,128) -> (128,64)
        int n = tid / CD, k = tid % CD;
        d1t[tid] = (bf16)d1[k * HID + n];
    }
    if (tid < D_MODEL * HID) {               // dec_w2 (128,768) -> (768,128)
        int n = tid / HID, k = tid % HID;
        d2t[tid] = (bf16)d2[k * D_MODEL + n];
    }
    if (tid == 0) *out_loss = 0.f;
}

// ---------------------------------------------------------------- encoder:
// z = GELU(x @ W1 + b1) @ W2 + b2, one 16-token tile per wave.
__global__ void __launch_bounds__(128) vq_encoder(const float* __restrict__ x,
        const float* __restrict__ b1, const float* __restrict__ b2,
        const bf16* __restrict__ w1t, const bf16* __restrict__ w2t,
        float* __restrict__ zf) {
    __shared__ __align__(32) bf16 hbuf[4][16 * 136];  // 136 = 128 + pad
    const int wave = threadIdx.x >> 5;
    const int lane = threadIdx.x & 31;
    const int cl   = lane & 15;
    const int hi   = lane >> 4;
    const int token0 = (blockIdx.x * 4 + wave) * 16;

    // ---- GEMM1: (16 x 768) x (768 x 128), f32 accum
    v8f acc[8];
#pragma unroll
    for (int n = 0; n < 8; n++) acc[n] = v8f_zero();

    const float* xrow = x + (size_t)(token0 + cl) * D_MODEL;
    for (int k0 = 0; k0 < D_MODEL; k0 += 32) {
        v16bf a = load_a_from_f32(xrow + k0 + hi * 8);
#pragma unroll
        for (int n = 0; n < 8; n++) {
            v16bf b = load_a_from_bf16(w1t + (size_t)(n * 16 + cl) * D_MODEL + k0 + hi * 8);
            acc[n] = wmma_bf16(a, b, acc[n]);
        }
    }
    // bias + exact GELU, stage h tile to LDS in bf16
#pragma unroll
    for (int n = 0; n < 8; n++) {
        float bias = b1[n * 16 + cl];
#pragma unroll
        for (int r = 0; r < 8; r++) {
            float v = gelu_exact(acc[n][r] + bias);
            hbuf[wave][(r + hi * 8) * 136 + n * 16 + cl] = (bf16)v;
        }
    }
    __syncthreads();

    // ---- GEMM2: (16 x 128) x (128 x 64)
    v16bf af[4];
#pragma unroll
    for (int kk = 0; kk < 4; kk++)
        af[kk] = load_a_from_bf16(&hbuf[wave][cl * 136 + kk * 32 + hi * 8]);

#pragma unroll
    for (int n = 0; n < 4; n++) {
        v8f zacc = v8f_zero();
#pragma unroll
        for (int kk = 0; kk < 4; kk++) {
            v16bf b = load_a_from_bf16(w2t + (size_t)(n * 16 + cl) * HID + kk * 32 + hi * 8);
            zacc = wmma_bf16(af[kk], b, zacc);
        }
        float bias = b2[n * 16 + cl];
#pragma unroll
        for (int r = 0; r < 8; r++)
            zf[(size_t)(token0 + r + hi * 8) * CD + n * 16 + cl] = zacc[r] + bias;
    }
}

// ---------------------------------------------------------------- quantize:
// argmin_c (||c||^2 - 2 z.c) via WMMA + shfl reduction; gather z_q; loss.
__global__ void __launch_bounds__(128) vq_quantize(const float* __restrict__ zf,
        const bf16* __restrict__ cbb, const float* __restrict__ cnorm,
        const float* __restrict__ cb32, bf16* __restrict__ zqb,
        float* __restrict__ out_idx, float* __restrict__ out_loss) {
    __shared__ int idxs[4][16];
    const int wave = threadIdx.x >> 5;
    const int lane = threadIdx.x & 31;
    const int cl   = lane & 15;
    const int hi   = lane >> 4;
    const int token0 = (blockIdx.x * 4 + wave) * 16;

    v16bf za[2];
#pragma unroll
    for (int kk = 0; kk < 2; kk++)
        za[kk] = load_a_from_f32(zf + (size_t)(token0 + cl) * CD + kk * 32 + hi * 8);

    float best[8]; int bidx[8];
#pragma unroll
    for (int r = 0; r < 8; r++) { best[r] = 3.4e38f; bidx[r] = 0; }

    for (int c0 = 0; c0 < NCODE; c0 += 16) {
        v8f acc = v8f_zero();
#pragma unroll
        for (int kk = 0; kk < 2; kk++) {
            v16bf b = load_a_from_bf16(cbb + (size_t)(c0 + cl) * CD + kk * 32 + hi * 8);
            acc = wmma_bf16(za[kk], b, acc);
        }
        float cn  = cnorm[c0 + cl];
        int code  = c0 + cl;
#pragma unroll
        for (int r = 0; r < 8; r++) {
            float s = cn - 2.0f * acc[r];
            if (s < best[r]) { best[r] = s; bidx[r] = code; }
        }
    }
    // reduce over the 16 lanes of each half-wave (row m lives in one half)
    for (int off = 1; off < 16; off <<= 1) {
#pragma unroll
        for (int r = 0; r < 8; r++) {
            float s2 = __shfl_xor(best[r], off, 32);
            int   i2 = __shfl_xor(bidx[r], off, 32);
            if (s2 < best[r] || (s2 == best[r] && i2 < bidx[r])) { best[r] = s2; bidx[r] = i2; }
        }
    }
    if (cl == 0) {
#pragma unroll
        for (int r = 0; r < 8; r++) {
            int m = r + hi * 8;
            idxs[wave][m] = bidx[r];
            out_idx[token0 + m] = (float)bidx[r];
        }
    }
    __syncthreads();

    // gather z_q (fp32 for loss, bf16 for decoder) + loss partial
    float lsum = 0.f;
    for (int e = lane; e < 16 * CD; e += 32) {
        int m = e >> 6, d = e & 63;
        int ci = idxs[wave][m];
        float q  = cb32[(size_t)ci * CD + d];
        float zv = zf[(size_t)(token0 + m) * CD + d];
        zqb[(size_t)(token0 + m) * CD + d] = (bf16)q;
        float df = q - zv;
        lsum += df * df;
    }
    for (int off = 16; off >= 1; off >>= 1) lsum += __shfl_xor(lsum, off, 32);
    if (lane == 0)
        atomicAdd(out_loss, lsum * (1.25f / (65536.0f * 64.0f)));
}

// ---------------------------------------------------------------- decoder:
// x_rec = GELU(z_q @ D1 + b1) @ D2 + b2, streamed to d_out.
__global__ void __launch_bounds__(128) vq_decoder(const bf16* __restrict__ zqb,
        const float* __restrict__ b1, const float* __restrict__ b2,
        const bf16* __restrict__ d1t, const bf16* __restrict__ d2t,
        float* __restrict__ xrec) {
    __shared__ __align__(32) bf16 hbuf[4][16 * 136];
    const int wave = threadIdx.x >> 5;
    const int lane = threadIdx.x & 31;
    const int cl   = lane & 15;
    const int hi   = lane >> 4;
    const int token0 = (blockIdx.x * 4 + wave) * 16;

    // ---- GEMM1: (16 x 64) x (64 x 128)
    v16bf za[2];
#pragma unroll
    for (int kk = 0; kk < 2; kk++)
        za[kk] = load_a_from_bf16(zqb + (size_t)(token0 + cl) * CD + kk * 32 + hi * 8);

    v8f acc[8];
#pragma unroll
    for (int n = 0; n < 8; n++) acc[n] = v8f_zero();
#pragma unroll
    for (int n = 0; n < 8; n++) {
#pragma unroll
        for (int kk = 0; kk < 2; kk++) {
            v16bf b = load_a_from_bf16(d1t + (size_t)(n * 16 + cl) * CD + kk * 32 + hi * 8);
            acc[n] = wmma_bf16(za[kk], b, acc[n]);
        }
    }
#pragma unroll
    for (int n = 0; n < 8; n++) {
        float bias = b1[n * 16 + cl];
#pragma unroll
        for (int r = 0; r < 8; r++) {
            float v = gelu_exact(acc[n][r] + bias);
            hbuf[wave][(r + hi * 8) * 136 + n * 16 + cl] = (bf16)v;
        }
    }
    __syncthreads();

    // ---- GEMM2: (16 x 128) x (128 x 768), 48 output tiles
    v16bf af[4];
#pragma unroll
    for (int kk = 0; kk < 4; kk++)
        af[kk] = load_a_from_bf16(&hbuf[wave][cl * 136 + kk * 32 + hi * 8]);

    for (int n2 = 0; n2 < 48; n2++) {
        v8f c = v8f_zero();
#pragma unroll
        for (int kk = 0; kk < 4; kk++) {
            v16bf b = load_a_from_bf16(d2t + (size_t)(n2 * 16 + cl) * HID + kk * 32 + hi * 8);
            c = wmma_bf16(af[kk], b, c);
        }
        float bias = b2[n2 * 16 + cl];
#pragma unroll
        for (int r = 0; r < 8; r++)
            xrec[(size_t)(token0 + r + hi * 8) * D_MODEL + n2 * 16 + cl] = c[r] + bias;
    }
}

// ---------------------------------------------------------------- launch

extern "C" void kernel_launch(void* const* d_in, const int* in_sizes, int n_in,
                              void* d_out, int out_size, void* d_ws, size_t ws_size,
                              hipStream_t stream) {
    const float* x      = (const float*)d_in[0];
    const float* enc_w1 = (const float*)d_in[1];
    const float* enc_b1 = (const float*)d_in[2];
    const float* enc_w2 = (const float*)d_in[3];
    const float* enc_b2 = (const float*)d_in[4];
    const float* cb     = (const float*)d_in[5];
    const float* dec_w1 = (const float*)d_in[6];
    const float* dec_b1 = (const float*)d_in[7];
    const float* dec_w2 = (const float*)d_in[8];
    const float* dec_b2 = (const float*)d_in[9];

    // workspace layout (all offsets 256B-aligned)
    char* ws = (char*)d_ws;
    float* zf   = (float*)(ws + 0);               // BN*64 fp32   (16 MB)
    bf16*  zqb  = (bf16*)(ws + 16777216);         // BN*64 bf16   (8 MB)
    bf16*  w1t  = (bf16*)(ws + 25165824);         // 128x768
    bf16*  w2t  = (bf16*)(ws + 25362432);         // 64x128
    bf16*  cbb  = (bf16*)(ws + 25378816);         // 2048x64
    float* cnorm= (float*)(ws + 25640960);        // 2048
    bf16*  d1t  = (bf16*)(ws + 25649152);         // 128x64
    bf16*  d2t  = (bf16*)(ws + 25665536);         // 768x128

    // output layout: x_rec (BN*768) | indices (BN) | vq_loss (1), all float
    float* out      = (float*)d_out;
    float* out_idx  = out + (size_t)BN * D_MODEL;
    float* out_loss = out + (size_t)BN * D_MODEL + BN;

    vq_prep<<<512, 256, 0, stream>>>(enc_w1, enc_w2, cb, dec_w1, dec_w2,
                                     w1t, w2t, cbb, cnorm, d1t, d2t, out_loss);
    vq_encoder<<<1024, 128, 0, stream>>>(x, enc_b1, enc_b2, w1t, w2t, zf);
    vq_quantize<<<1024, 128, 0, stream>>>(zf, cbb, cnorm, cb, zqb, out_idx, out_loss);
    vq_decoder<<<1024, 128, 0, stream>>>(zqb, dec_b1, dec_b2, d1t, d2t, out);
}
